// GenLayer_68478958568058
// MI455X (gfx1250) — matrix operand order repaired and checked by reading, not compiled, not run on hardware
//
#include <hip/hip_runtime.h>

#define DD 512
#define BB 64
#define SS 256
#define TT 128

typedef __bf16 bf16_t;
typedef __attribute__((ext_vector_type(16))) __bf16 v16bf;
typedef __attribute__((ext_vector_type(8)))  __bf16 v8bf;
typedef __attribute__((ext_vector_type(8)))  float  v8f;

// ---------------- workspace layout (bytes) ----------------
static constexpr size_t OFF_WT   = 0;                               // 11 * D*D bf16 (transposed, N-major)
static constexpr size_t SZ_WT    = 11ull * DD * DD * 2;
static constexpr size_t OFF_K2   = OFF_WT + SZ_WT;                  // [B][S][D] f32
static constexpr size_t SZ_KV2   = (size_t)BB * SS * DD * 4;
static constexpr size_t OFF_V2   = OFF_K2 + SZ_KV2;
static constexpr size_t OFF_K1   = OFF_V2 + SZ_KV2;                 // [4g][8h][16b][128pos][64] f32
static constexpr size_t SZ_K1    = 32ull * 16 * 128 * 64 * 4;
static constexpr size_t OFF_V1   = OFF_K1 + SZ_K1;
static constexpr size_t OFF_OB   = OFF_V1 + SZ_K1;                  // [4g][16b][512] f32
static constexpr size_t OFF_PART = OFF_OB   + 4ull * 16 * 512 * 4;  // [4g][16b][512] f32
static constexpr size_t OFF_PE   = OFF_PART + 4ull * 16 * 512 * 4;  // [128][512] f32
static constexpr size_t OFF_SYNC = OFF_PE + (size_t)TT * DD * 4;    // 64 u32

// ---------------- WMMA helpers ----------------
// One 16x16x32 bf16 WMMA k-step. A: f32 row pointer (LDS or global), converted on the fly.
// B: bf16 transposed-weight row (contiguous K).
__device__ __forceinline__ v8f wmma_kstep(const float* __restrict__ arow,
                                          const bf16_t* __restrict__ brow,
                                          int kbase, int lane, v8f acc) {
  const int kh = lane >> 4;
  const int ka = kbase + kh * 8;    // A: interleaved 8-element chunks (ISA dense A layout)
  const int kb = kbase + kh * 8;    // B: interleaved 8-element chunks (matches A)
  const float4 a0 = *(const float4*)(arow + ka);
  const float4 a1 = *(const float4*)(arow + ka + 4);
  const float4 a2 = *(const float4*)(arow + ka + 16);
  const float4 a3 = *(const float4*)(arow + ka + 20);
  v16bf av, bv;
  av[0]=(bf16_t)a0.x; av[1]=(bf16_t)a0.y; av[2]=(bf16_t)a0.z; av[3]=(bf16_t)a0.w;
  av[4]=(bf16_t)a1.x; av[5]=(bf16_t)a1.y; av[6]=(bf16_t)a1.z; av[7]=(bf16_t)a1.w;
  av[8]=(bf16_t)a2.x; av[9]=(bf16_t)a2.y; av[10]=(bf16_t)a2.z; av[11]=(bf16_t)a2.w;
  av[12]=(bf16_t)a3.x; av[13]=(bf16_t)a3.y; av[14]=(bf16_t)a3.z; av[15]=(bf16_t)a3.w;
  const v8bf b0 = *(const v8bf*)(brow + kb);
  const v8bf b1 = *(const v8bf*)(brow + kb + 16);
#pragma unroll
  for (int j = 0; j < 8; ++j) { bv[j] = b0[j]; bv[8 + j] = b1[j]; }
  return __builtin_amdgcn_wmma_f32_16x16x32_bf16(false, av, false, bv, (short)0, acc, false, false);
}

// Y[16][64] = A[16][512] @ Wt_slice^T + bias.  512 threads: wave = kq*4 + nt.
// Each of 16 waves does one (n-tile, k-quarter); partials reduced through LDS.
__device__ __forceinline__ void gemm16x64(const float* __restrict__ A, int astride,
                                          const bf16_t* __restrict__ wt,   // [64 cols][512 k] bf16
                                          const float* __restrict__ bias,  // 64
                                          float* __restrict__ ybuf,        // LDS [16*64]
                                          float* __restrict__ red) {       // LDS >= 16*256 f32
  const int tid = threadIdx.x, wave = tid >> 5, lane = tid & 31;
  const int nt = wave & 3, kq = wave >> 2;
  const int mr = lane & 15;
  const float*  arow = A + (size_t)mr * astride;
  const bf16_t* brow = wt + (size_t)(nt * 16 + mr) * DD;
  v8f acc = {};
#pragma unroll
  for (int ks = 0; ks < 4; ++ks)
    acc = wmma_kstep(arow, brow, kq * 128 + ks * 32, lane, acc);
  float* rb = red + (wave * 256 + lane * 8);
#pragma unroll
  for (int r = 0; r < 8; ++r) rb[r] = acc[r];
  __syncthreads();
#pragma unroll
  for (int e = 0; e < 2; ++e) {
    int o = tid * 2 + e;              // o = m*64 + n
    int n = o & 63, m = o >> 6;
    int ntile = n >> 4, nn = n & 15;
    int lsrc = nn + ((m & 8) ? 16 : 0);
    int r = m & 7;
    float s = 0.f;
#pragma unroll
    for (int q = 0; q < 4; ++q) s += red[(q * 4 + ntile) * 256 + lsrc * 8 + r];
    ybuf[o] = s + bias[n];
  }
  __syncthreads();
}

// ---------------- precompute kernels ----------------
struct WtArgs { const float* src[11]; unsigned short* dst; };

__global__ void wt_kernel(WtArgs a) {
  size_t idx = (size_t)blockIdx.x * 256 + threadIdx.x;
  if (idx >= 11ull * DD * DD) return;
  int mat = (int)(idx / (DD * DD));
  int rem = (int)(idx % (DD * DD));
  int n = rem / DD, k = rem % DD;
  bf16_t v = (bf16_t)a.src[mat][(size_t)k * DD + n];   // Wt[n][k] = W[k][n]
  a.dst[idx] = __builtin_bit_cast(unsigned short, v);
}

__global__ void pe_kernel(float* pe) {
  int idx = blockIdx.x * 256 + threadIdx.x;             // 128*512
  int i = idx >> 9, d = idx & 511;
  int j = d >> 1;
  float div = __expf((float)(2 * j) * (-9.210340371976184f / 512.0f));
  float ang = (float)i * div;
  pe[idx] = (d & 1) ? __cosf(ang) : __sinf(ang);
}

__global__ void init_kernel(unsigned* sync) { sync[threadIdx.x] = 0u; }

struct KvArgs {
  const float* hc; const unsigned short* wtk; const unsigned short* wtv;
  const float* bk; const float* bv; float* k2; float* v2;
};

__global__ __launch_bounds__(256) void kv2_kernel(KvArgs a) {
  const int rt = blockIdx.x;                 // 1024 row-tiles of B*S
  const int tid = threadIdx.x, wave = tid >> 5, lane = tid & 31;
  const int mr = lane & 15;
  const float* arow = a.hc + ((size_t)rt * 16 + mr) * DD;
  const bf16_t* wk = reinterpret_cast<const bf16_t*>(a.wtk);
  const bf16_t* wv = reinterpret_cast<const bf16_t*>(a.wtv);
#pragma unroll 1
  for (int j = 0; j < 8; ++j) {
    int job = wave + j * 8;                  // 0..63 : mat(2) x ntile(32)
    int mat = job >> 5, nt = job & 31;
    const bf16_t* brow = (mat ? wv : wk) + (size_t)(nt * 16 + mr) * DD;
    const float* bias = mat ? a.bv : a.bk;
    float* dst = mat ? a.v2 : a.k2;
    v8f acc = {};
#pragma unroll
    for (int ks = 0; ks < 16; ++ks)
      acc = wmma_kstep(arow, brow, ks * 32, lane, acc);
    int nn = nt * 16 + (lane & 15);
    int mbase = (lane >> 4) * 8;
    float bvv = bias[nn];
#pragma unroll
    for (int r = 0; r < 8; ++r)
      dst[((size_t)rt * 16 + mbase + r) * DD + nn] = acc[r] + bvv;
  }
}

// ---------------- decode (persistent) ----------------
struct DecArgs {
  const float *noise, *w0;
  const float *bq1, *bk1, *bv1, *bo1, *bq2, *bo2, *b1, *b2, *b3;
  const float *ln1g, *ln1b, *ln2g, *ln2b, *lnpg, *lnpb;
  const unsigned short* wt;        // 11 transposed bf16 matrices
  const float *k2, *v2;            // [B][S][D]
  float *k1, *v1;                  // [32 wg][16][128][64]
  float *obuf, *part;              // [4][16][512]
  const float* pe;                 // [128][512]
  unsigned* sync;                  // [4*16]
  float *out_w, *out_z, *out_wh, *out_mean, *out_std;
};

__device__ __forceinline__ void group_sync(unsigned* cnt, unsigned target) {
  __threadfence();                 // release our stores (every thread/wave)
  __syncthreads();
  if (threadIdx.x == 0) {
    __hip_atomic_fetch_add(cnt, 1u, __ATOMIC_RELAXED, __HIP_MEMORY_SCOPE_AGENT);
    unsigned v;
    do {
      __builtin_amdgcn_s_sleep(2);
      v = __hip_atomic_load(cnt, __ATOMIC_RELAXED, __HIP_MEMORY_SCOPE_AGENT);
    } while (v < target);
  }
  __syncthreads();
  __threadfence();                 // acquire (invalidate) on every wave/CU
}

// one attention query per wave (b = wave). Head-sliced K/V rows of 64 f32.
template <int NP>
__device__ __forceinline__ void attn_one(const float* __restrict__ qb,     // LDS [16][64]
                                         const float* __restrict__ Kbase,  // + b*bstride + pos*pstride
                                         const float* __restrict__ Vbase,
                                         size_t bstride, int pstride, int P,
                                         float* __restrict__ sbuf,         // LDS [16][260]
                                         float* __restrict__ outp, int ostride) {
  const int tid = threadIdx.x, b = tid >> 5, lane = tid & 31;
  const float* q = qb + b * 64;
  float4 qv[16];
#pragma unroll
  for (int c = 0; c < 16; ++c) qv[c] = *(const float4*)(q + 4 * c);
  const float* Kb = Kbase + (size_t)b * bstride;
  const float* Vb = Vbase + (size_t)b * bstride;
  float* srow = sbuf + b * 260;
  float sc[NP];
  float mx = -3.0e38f;
#pragma unroll
  for (int p = 0; p < NP; ++p) {
    int pos = lane + (p << 5);
    float d = -3.0e38f;
    if (pos < P) {
      const float* kr = Kb + (size_t)pos * pstride;
      float acc = 0.f;
#pragma unroll
      for (int c = 0; c < 16; ++c) {
        float4 kv = *(const float4*)(kr + 4 * c);
        acc += qv[c].x * kv.x + qv[c].y * kv.y + qv[c].z * kv.z + qv[c].w * kv.w;
      }
      d = acc * 0.125f;            // 1/sqrt(64)
    }
    sc[p] = d;
    mx = fmaxf(mx, d);
  }
#pragma unroll
  for (int o = 16; o > 0; o >>= 1) mx = fmaxf(mx, __shfl_xor(mx, o, 32));
  float sum = 0.f;
#pragma unroll
  for (int p = 0; p < NP; ++p) {
    int pos = lane + (p << 5);
    if (pos < P) { float e = __expf(sc[p] - mx); srow[pos] = e; sum += e; }
  }
#pragma unroll
  for (int o = 16; o > 0; o >>= 1) sum += __shfl_xor(sum, o, 32);
  float rs = 1.0f / sum;
  float o0 = 0.f, o1 = 0.f;
  for (int pos = 0; pos < P; ++pos) {
    float pw = srow[pos];
    const float* vr = Vb + (size_t)pos * pstride;
    o0 += pw * vr[lane];
    o1 += pw * vr[lane + 32];
  }
  float* orow = outp + (size_t)b * ostride;
  orow[lane] = o0 * rs;
  orow[lane + 32] = o1 * rs;
}

__device__ __forceinline__ void ln_row(float* __restrict__ x,               // LDS row
                                       const float* __restrict__ resid,     // global row (512)
                                       const float* __restrict__ pe,        // nullable
                                       const float* __restrict__ gg,
                                       const float* __restrict__ bb, int lane) {
#pragma unroll
  for (int c = lane; c < 512; c += 32) {
    float v = x[c] + resid[c];
    if (pe) v += pe[c];
    x[c] = v;
  }
  float s = 0.f;
#pragma unroll
  for (int c = lane; c < 512; c += 32) s += x[c];
#pragma unroll
  for (int o = 16; o > 0; o >>= 1) s += __shfl_xor(s, o, 32);
  float mean = s * (1.0f / 512.0f);
  float v2 = 0.f;
#pragma unroll
  for (int c = lane; c < 512; c += 32) { float d = x[c] - mean; v2 += d * d; }
#pragma unroll
  for (int o = 16; o > 0; o >>= 1) v2 += __shfl_xor(v2, o, 32);
  float rsd = rsqrtf(v2 * (1.0f / 512.0f) + 1e-5f);
#pragma unroll
  for (int c = lane; c < 512; c += 32) x[c] = (x[c] - mean) * rsd * gg[c] + bb[c];
}

__global__ __launch_bounds__(512, 1) void decode_kernel(DecArgs a) {
  __shared__ float xf[16 * 520];       // activation rows (padded stride)
  __shared__ float ybuf[16 * 64];      // GEMM slice output
  __shared__ float qbuf[16 * 64];      // q slice / mean
  __shared__ float scratch[16 * 260];  // GEMM reduce / softmax probs

  const int g = blockIdx.x >> 3, h = blockIdx.x & 7;
  const int tid = threadIdx.x, wave = tid >> 5, lane = tid & 31;
  unsigned* cnt = a.sync + g * 16;
  unsigned epoch = 0;

  const bf16_t* wtb = reinterpret_cast<const bf16_t*>(a.wt);
  const bf16_t* w_q1 = wtb + ((size_t)0 * DD + h * 64) * DD;
  const bf16_t* w_k1 = wtb + ((size_t)1 * DD + h * 64) * DD;
  const bf16_t* w_v1 = wtb + ((size_t)2 * DD + h * 64) * DD;
  const bf16_t* w_o1 = wtb + ((size_t)3 * DD + h * 64) * DD;
  const bf16_t* w_q2 = wtb + ((size_t)4 * DD + h * 64) * DD;
  const bf16_t* w_o2 = wtb + ((size_t)5 * DD + h * 64) * DD;
  const bf16_t* w_w1 = wtb + ((size_t)6 * DD + h * 64) * DD;
  const bf16_t* w_w2 = wtb + ((size_t)7 * DD + h * 64) * DD;
  const bf16_t* w_w3 = wtb + ((size_t)8 * DD + h * 64) * DD;

  float* k1g = a.k1 + (size_t)(g * 8 + h) * 16 * 128 * 64;   // [16b][128][64]
  float* v1g = a.v1 + (size_t)(g * 8 + h) * 16 * 128 * 64;
  const float* k2g = a.k2 + (size_t)(g * 16) * SS * DD + h * 64;  // + b*S*D + pos*D
  const float* v2g = a.v2 + (size_t)(g * 16) * SS * DD + h * 64;
  float* ob  = a.obuf + (size_t)g * 16 * 512;
  float* pb  = a.part + (size_t)g * 16 * 512;

  // x = w0 (broadcast over batch rows)
  {
    float w0v = a.w0[tid];
#pragma unroll
    for (int m = 0; m < 16; ++m) xf[m * 520 + tid] = w0v;
  }
  __syncthreads();

#pragma unroll 1
  for (int i = 0; i < TT; ++i) {
    // ---- MHA1: q/k/v head-slice projections ----
    gemm16x64(xf, 520, w_q1, a.bq1 + h * 64, ybuf, scratch);
    qbuf[tid] = ybuf[tid]; qbuf[tid + 512] = ybuf[tid + 512];
    __syncthreads();
    gemm16x64(xf, 520, w_k1, a.bk1 + h * 64, ybuf, scratch);
#pragma unroll
    for (int e = 0; e < 2; ++e) {
      int o = tid + (e << 9); int b = o >> 6, dd = o & 63;
      k1g[((size_t)b * 128 + i) * 64 + dd] = ybuf[o];
    }
    __syncthreads();
    gemm16x64(xf, 520, w_v1, a.bv1 + h * 64, ybuf, scratch);
#pragma unroll
    for (int e = 0; e < 2; ++e) {
      int o = tid + (e << 9); int b = o >> 6, dd = o & 63;
      v1g[((size_t)b * 128 + i) * 64 + dd] = ybuf[o];
    }
    __threadfence_block();
    __syncthreads();
    // causal self-attention over prefix 0..i
    attn_one<4>(qbuf, k1g, v1g, (size_t)128 * 64, 64, i + 1, scratch, ob + h * 64, 512);
    ++epoch; group_sync(cnt, epoch * 8);                       // o1 assembled

    // ---- wo1 slice + residual + LN1 ----
    gemm16x64(ob, 512, w_o1, a.bo1 + h * 64, ybuf, scratch);
#pragma unroll
    for (int e = 0; e < 2; ++e) {
      int o = tid + (e << 9); int b = o >> 6, dd = o & 63;
      pb[b * 512 + h * 64 + dd] = ybuf[o];
    }
    ++epoch; group_sync(cnt, epoch * 8);                       // wo1 assembled
    ln_row(xf + wave * 520, pb + wave * 512, nullptr, a.ln1g, a.ln1b, lane);
    __syncthreads();

    // ---- MHA2 (cross over h_C, K2/V2 precomputed) ----
    gemm16x64(xf, 520, w_q2, a.bq2 + h * 64, ybuf, scratch);
    qbuf[tid] = ybuf[tid]; qbuf[tid + 512] = ybuf[tid + 512];
    __syncthreads();
    attn_one<8>(qbuf, k2g, v2g, (size_t)SS * DD, DD, SS, scratch, ob + h * 64, 512);
    ++epoch; group_sync(cnt, epoch * 8);                       // o2 assembled
    gemm16x64(ob, 512, w_o2, a.bo2 + h * 64, ybuf, scratch);
#pragma unroll
    for (int e = 0; e < 2; ++e) {
      int o = tid + (e << 9); int b = o >> 6, dd = o & 63;
      pb[b * 512 + h * 64 + dd] = ybuf[o];
    }
    ++epoch; group_sync(cnt, epoch * 8);                       // wo2 assembled
    ln_row(xf + wave * 520, pb + wave * 512, nullptr, a.ln2g, a.ln2b, lane);
    __syncthreads();
    // xf = w_hat : emit slice
#pragma unroll
    for (int e = 0; e < 2; ++e) {
      int o = tid + (e << 9); int b = o >> 6, dd = o & 63;
      size_t oidx = ((size_t)(g * 16 + b) * TT + i) * DD + h * 64 + dd;
      a.out_wh[oidx] = xf[b * 520 + h * 64 + dd];
    }

    // ---- mean / std / z ----
    gemm16x64(xf, 520, w_w1, a.b1 + h * 64, ybuf, scratch);
    qbuf[tid] = ybuf[tid]; qbuf[tid + 512] = ybuf[tid + 512];  // keep mean
    __syncthreads();
    gemm16x64(xf, 520, w_w2, a.b2 + h * 64, ybuf, scratch);
#pragma unroll
    for (int e = 0; e < 2; ++e) {
      int o = tid + (e << 9); int b = o >> 6, dd = o & 63;
      int col = h * 64 + dd;
      size_t oidx = ((size_t)(g * 16 + b) * TT + i) * DD + col;
      float stdv = __expf(ybuf[o]);
      float meanv = qbuf[o];
      float z = meanv + a.noise[oidx] * stdv;
      a.out_mean[oidx] = meanv;
      a.out_std[oidx]  = stdv;
      a.out_z[oidx]    = z;
      ob[b * 512 + col] = z;
    }
    ++epoch; group_sync(cnt, epoch * 8);                       // z assembled

    // ---- z @ w3 + residual + pe + LNp -> w_new ----
    gemm16x64(ob, 512, w_w3, a.b3 + h * 64, ybuf, scratch);
#pragma unroll
    for (int e = 0; e < 2; ++e) {
      int o = tid + (e << 9); int b = o >> 6, dd = o & 63;
      pb[b * 512 + h * 64 + dd] = ybuf[o];
    }
    ++epoch; group_sync(cnt, epoch * 8);                       // w3 assembled
    ln_row(xf + wave * 520, pb + wave * 512, a.pe + (size_t)i * DD, a.lnpg, a.lnpb, lane);
    __syncthreads();
#pragma unroll
    for (int e = 0; e < 2; ++e) {
      int o = tid + (e << 9); int b = o >> 6, dd = o & 63;
      size_t oidx = ((size_t)(g * 16 + b) * TT + i) * DD + h * 64 + dd;
      a.out_w[oidx] = xf[b * 520 + h * 64 + dd];
    }
    // xf now holds w_new for step i+1
  }
}

// ---------------- launcher ----------------
extern "C" void kernel_launch(void* const* d_in, const int* in_sizes, int n_in,
                              void* d_out, int out_size, void* d_ws, size_t ws_size,
                              hipStream_t stream) {
  (void)in_sizes; (void)n_in; (void)out_size; (void)ws_size;
  auto F = [&](int i) { return (const float*)d_in[i]; };
  // 0 h_C, 1 noise, 2 w0,
  // 3 wq1 4 bq1 5 wk1 6 bk1 7 wv1 8 bv1 9 wo1 10 bo1,
  // 11 wq2 12 bq2 13 wk2 14 bk2 15 wv2 16 bv2 17 wo2 18 bo2,
  // 19 w1 20 b1 21 w2 22 b2 23 w3 24 b3,
  // 25 ln1g 26 ln1b 27 ln2g 28 ln2b 29 lnpg 30 lnpb
  char* ws = (char*)d_ws;
  unsigned short* wt = (unsigned short*)(ws + OFF_WT);
  float* k2 = (float*)(ws + OFF_K2);
  float* v2 = (float*)(ws + OFF_V2);
  float* k1 = (float*)(ws + OFF_K1);
  float* v1 = (float*)(ws + OFF_V1);
  float* ob = (float*)(ws + OFF_OB);
  float* pb = (float*)(ws + OFF_PART);
  float* pe = (float*)(ws + OFF_PE);
  unsigned* sync = (unsigned*)(ws + OFF_SYNC);

  // 1) weight transpose -> bf16 (N-major): q1,k1,v1,o1,q2,o2,w1,w2,w3,k2,v2
  WtArgs wa;
  const int src_idx[11] = {3, 5, 7, 9, 11, 17, 19, 21, 23, 13, 15};
  for (int m = 0; m < 11; ++m) wa.src[m] = F(src_idx[m]);
  wa.dst = wt;
  wt_kernel<<<(int)((11ull * DD * DD + 255) / 256), 256, 0, stream>>>(wa);

  // 2) positional-encoding table
  pe_kernel<<<(TT * DD) / 256, 256, 0, stream>>>(pe);

  // 3) reset group-sync counters (required each launch for graph replay)
  init_kernel<<<1, 64, 0, stream>>>(sync);

  // 4) cross-attention K2/V2 = h_C @ {wk2,wv2} + bias  (WMMA GEMM)
  KvArgs kv;
  kv.hc = F(0);
  kv.wtk = wt + (size_t)9 * DD * DD;
  kv.wtv = wt + (size_t)10 * DD * DD;
  kv.bk = F(14); kv.bv = F(16);
  kv.k2 = k2; kv.v2 = v2;
  kv2_kernel<<<(BB * SS) / 16, 256, 0, stream>>>(kv);

  // 5) persistent decode: 32 WGs = 4 batch-chunks x 8 heads
  const size_t N = (size_t)BB * TT * DD;
  float* outb = (float*)d_out;
  DecArgs da;
  da.noise = F(1); da.w0 = F(2);
  da.bq1 = F(4); da.bk1 = F(6); da.bv1 = F(8); da.bo1 = F(10);
  da.bq2 = F(12); da.bo2 = F(18);
  da.b1 = F(20); da.b2 = F(22); da.b3 = F(24);
  da.ln1g = F(25); da.ln1b = F(26); da.ln2g = F(27); da.ln2b = F(28);
  da.lnpg = F(29); da.lnpb = F(30);
  da.wt = wt; da.k2 = k2; da.v2 = v2; da.k1 = k1; da.v1 = v1;
  da.obuf = ob; da.part = pb; da.pe = pe; da.sync = sync;
  da.out_w = outb; da.out_z = outb + N; da.out_wh = outb + 2 * N;
  da.out_mean = outb + 3 * N; da.out_std = outb + 4 * N;
  decode_kernel<<<32, 512, 0, stream>>>(da);
}
